// GraphModule_65197603553449
// MI455X (gfx1250) — matrix operand order repaired and checked by reading, not compiled, not run on hardware
//
#include <hip/hip_runtime.h>
#include <hip/hip_bf16.h>

typedef __attribute__((ext_vector_type(16))) _Float16 v16h;
typedef __attribute__((ext_vector_type(8)))  _Float16 v8h;
typedef __attribute__((ext_vector_type(4)))  _Float16 v4h;
typedef __attribute__((ext_vector_type(8)))  float    v8f;

union V16 { v16h v; v8h h[2]; };

#define NN    100000
#define EE    300000
#define DIN_  128
#define HH    512
#define OUTC  256
#define GG    4096
#define EPS_  1e-5f
#define SLOPE_ 0.01f
#define SELFW 2.0f

// ---------------- init kernels ----------------
__global__ void k_init_deg(float* deg) {
    int i = blockIdx.x * blockDim.x + threadIdx.x;
    if (i < NN) deg[i] = SELFW;
}
__global__ void k_zero_stats(float* stats) {
    int i = blockIdx.x * blockDim.x + threadIdx.x;
    if (i < 4 * HH) stats[i] = 0.0f;
}
__global__ void k_init_pooled(float* pooled) {
    long i = (long)blockIdx.x * blockDim.x + threadIdx.x;
    if (i < (long)GG * HH) pooled[i] = -__builtin_inff();
}

// ---------------- degree / dinv ----------------
__global__ void k_deg_edges(const int* __restrict__ col, float* __restrict__ deg) {
    int e = blockIdx.x * blockDim.x + threadIdx.x;
    if (e < EE) atomicAdd(&deg[col[e]], 1.0f);
}
__global__ void k_dinv(float* deg) {
    int i = blockIdx.x * blockDim.x + threadIdx.x;
    if (i < NN) deg[i] = rsqrtf(deg[i]);
}

// ---------------- split-f16 WMMA GEMM ----------------
// C[M,Nc] = A[M,K] @ B[K,Nc] (+bias), near-f32 accuracy via hi/lo f16 split
// (ah*bh + ah*bl + al*bh, f32 accumulate).
// Block tile: 32(M) x 256(N). 8 waves, each computes 32x32 (2x2 WMMA tiles).
// hi/lo split done ONCE at staging (float4 global loads); inner loop is pure
// ds_load_b128 + v_wmma. Requires M%32==0, Nc%256==0, K%32==0 (all call sites).
__launch_bounds__(256)
__global__ void k_gemm(const float* __restrict__ A, const float* __restrict__ B,
                       const float* __restrict__ bias, float* __restrict__ C,
                       int M, int K, int Nc, int sanitizeA)
{
    __shared__ _Float16 sAh[32 * 32], sAl[32 * 32];        // [m][k]
    __shared__ _Float16 sBh[256 * 32], sBl[256 * 32];      // transposed: [n][k]

    const int tid  = threadIdx.x;
    const int lane = tid & 31;
    const int w    = tid >> 5;       // wave 0..7 -> N sub-slab w*32
    const int l    = lane & 15;
    const int hb   = lane >> 4;      // half-wave (0/1)

    const int nBlocksN = Nc >> 8;    // 2 (Nc=512) or 1 (Nc=256)
    const int bm   = blockIdx.x / nBlocksN;
    const int bnb  = blockIdx.x % nBlocksN;
    const int row0 = bm * 32;
    const int colB = bnb * 256;      // block N origin
    const int col0 = colB + w * 32;  // wave N origin

    // staging coordinates (float4 granularity)
    const int aM = tid >> 3;          // 0..31
    const int aK = (tid & 7) * 4;     // 0,4,...,28

    v8f acc00 = {}, acc01 = {}, acc10 = {}, acc11 = {};

    for (int k0 = 0; k0 < K; k0 += 32) {
        // ---- stage A 32x32: one float4 per thread (global b128, coalesced) ----
        {
            float4 v = *(const float4*)&A[(long)(row0 + aM) * K + (k0 + aK)];
            if (sanitizeA) {
                if (!__builtin_isfinite(v.x)) v.x = 0.0f;   // -inf pooled -> 0
                if (!__builtin_isfinite(v.y)) v.y = 0.0f;
                if (!__builtin_isfinite(v.z)) v.z = 0.0f;
                if (!__builtin_isfinite(v.w)) v.w = 0.0f;
            }
            v4h hi4, lo4;
            hi4[0] = (_Float16)v.x; lo4[0] = (_Float16)(v.x - (float)hi4[0]);
            hi4[1] = (_Float16)v.y; lo4[1] = (_Float16)(v.y - (float)hi4[1]);
            hi4[2] = (_Float16)v.z; lo4[2] = (_Float16)(v.z - (float)hi4[2]);
            hi4[3] = (_Float16)v.w; lo4[3] = (_Float16)(v.w - (float)hi4[3]);
            *(v4h*)&sAh[aM * 32 + aK] = hi4;
            *(v4h*)&sAl[aM * 32 + aK] = lo4;
        }
        // ---- stage B 32x256: 8 float4/thread (coalesced), transpose+split ----
        #pragma unroll
        for (int r = 0; r < 8; ++r) {
            int i   = (tid + r * 256) * 4;   // element 0..8188 step 4
            int bk  = i >> 8;
            int bnn = i & 255;
            float4 v = *(const float4*)&B[(long)(k0 + bk) * Nc + (colB + bnn)];
            _Float16 h0 = (_Float16)v.x, h1 = (_Float16)v.y;
            _Float16 h2 = (_Float16)v.z, h3 = (_Float16)v.w;
            sBh[(bnn + 0) * 32 + bk] = h0;
            sBh[(bnn + 1) * 32 + bk] = h1;
            sBh[(bnn + 2) * 32 + bk] = h2;
            sBh[(bnn + 3) * 32 + bk] = h3;
            sBl[(bnn + 0) * 32 + bk] = (_Float16)(v.x - (float)h0);
            sBl[(bnn + 1) * 32 + bk] = (_Float16)(v.y - (float)h1);
            sBl[(bnn + 2) * 32 + bk] = (_Float16)(v.z - (float)h2);
            sBl[(bnn + 3) * 32 + bk] = (_Float16)(v.w - (float)h3);
        }
        // prefetch next K-slab while this one is consumed
        if (k0 + 32 < K) {
            __builtin_prefetch(&A[(long)(row0 + aM) * K + (k0 + 32 + aK)], 0, 3);
            __builtin_prefetch(&B[(long)(k0 + 32) * Nc + (colB + (tid & 255))], 0, 3);
        }
        __syncthreads();

        // ---- load WMMA fragments (pure b128 LDS loads, no conversions) ----
        // A k order used here (contiguous, not ISA table order): lane half hb: halfs 0..15 -> K=hb*16+j
        // (same k order as B)  => one contiguous 16-half run.
        V16 ah0, al0, ah1, al1;
        {
            int ro0 = l * 32 + hb * 16;          // M-tile 0, row l
            int ro1 = (16 + l) * 32 + hb * 16;   // M-tile 1
            ah0.h[0] = *(const v8h*)&sAh[ro0];
            ah0.h[1] = *(const v8h*)&sAh[ro0 + 8];
            al0.h[0] = *(const v8h*)&sAl[ro0];
            al0.h[1] = *(const v8h*)&sAl[ro0 + 8];
            ah1.h[0] = *(const v8h*)&sAh[ro1];
            ah1.h[1] = *(const v8h*)&sAh[ro1 + 8];
            al1.h[0] = *(const v8h*)&sAl[ro1];
            al1.h[1] = *(const v8h*)&sAl[ro1 + 8];
        }
        // B k order used here (contiguous, matches A): half t -> K = hb*16+t, N = l.
        // With sB transposed [n][k] this is one contiguous 16-half run.
        const int bo0 = (w * 32 + l) * 32 + hb * 16;        // N-tile 0
        const int bo1 = (w * 32 + 16 + l) * 32 + hb * 16;   // N-tile 1
        v16h bh0 = *(const v16h*)&sBh[bo0];
        v16h bl0 = *(const v16h*)&sBl[bo0];
        v16h bh1 = *(const v16h*)&sBh[bo1];
        v16h bl1 = *(const v16h*)&sBl[bo1];

        // ---- 12 WMMAs, interleaved across accumulators ----
        acc00 = __builtin_amdgcn_wmma_f32_16x16x32_f16(false, ah0.v, false, bh0, (short)0, acc00, false, false);
        acc01 = __builtin_amdgcn_wmma_f32_16x16x32_f16(false, ah0.v, false, bh1, (short)0, acc01, false, false);
        acc10 = __builtin_amdgcn_wmma_f32_16x16x32_f16(false, ah1.v, false, bh0, (short)0, acc10, false, false);
        acc11 = __builtin_amdgcn_wmma_f32_16x16x32_f16(false, ah1.v, false, bh1, (short)0, acc11, false, false);
        acc00 = __builtin_amdgcn_wmma_f32_16x16x32_f16(false, ah0.v, false, bl0, (short)0, acc00, false, false);
        acc01 = __builtin_amdgcn_wmma_f32_16x16x32_f16(false, ah0.v, false, bl1, (short)0, acc01, false, false);
        acc10 = __builtin_amdgcn_wmma_f32_16x16x32_f16(false, ah1.v, false, bl0, (short)0, acc10, false, false);
        acc11 = __builtin_amdgcn_wmma_f32_16x16x32_f16(false, ah1.v, false, bl1, (short)0, acc11, false, false);
        acc00 = __builtin_amdgcn_wmma_f32_16x16x32_f16(false, al0.v, false, bh0, (short)0, acc00, false, false);
        acc01 = __builtin_amdgcn_wmma_f32_16x16x32_f16(false, al0.v, false, bh1, (short)0, acc01, false, false);
        acc10 = __builtin_amdgcn_wmma_f32_16x16x32_f16(false, al1.v, false, bh0, (short)0, acc10, false, false);
        acc11 = __builtin_amdgcn_wmma_f32_16x16x32_f16(false, al1.v, false, bh1, (short)0, acc11, false, false);
        __syncthreads();
    }

    // ---- store: C/D f32 layout: VGPR r -> M = base + hb*8 + r, N = base + l ----
    float bv0 = bias ? bias[col0 + l] : 0.0f;
    float bv1 = bias ? bias[col0 + 16 + l] : 0.0f;
    #pragma unroll
    for (int r = 0; r < 8; ++r) {
        long m0 = (long)(row0 + hb * 8 + r);
        long m1 = (long)(row0 + 16 + hb * 8 + r);
        C[m0 * Nc + (col0 + l)]      = acc00[r] + bv0;
        C[m0 * Nc + (col0 + 16 + l)] = acc01[r] + bv1;
        C[m1 * Nc + (col0 + l)]      = acc10[r] + bv0;
        C[m1 * Nc + (col0 + 16 + l)] = acc11[r] + bv1;
    }
}

// ---------------- GCN aggregation (float4 per thread) ----------------
// out[i,c] = h[i,c] * (SELFW * dinv[i]^2) + bias[c]   (self-loop message + bias)
__global__ void k_selfloop_bias(const float* __restrict__ h, const float* __restrict__ dinv,
                                const float* __restrict__ bias, float* __restrict__ out)
{
    long g4 = (long)blockIdx.x * blockDim.x + threadIdx.x;
    if (g4 >= (long)NN * HH / 4) return;
    long base = g4 * 4;
    long node = base / HH;
    int  c    = (int)(base - node * HH);     // multiple of 4; HH%4==0 -> no row crossing
    float di  = dinv[node];
    float s   = SELFW * di * di;
    float4 hv = *(const float4*)&h[base];
    float4 bv = *(const float4*)&bias[c];
    float4 o;
    o.x = hv.x * s + bv.x;
    o.y = hv.y * s + bv.y;
    o.z = hv.z * s + bv.z;
    o.w = hv.w * s + bv.w;
    *(float4*)&out[base] = o;
}

// scatter-add edge messages: out[col] += h[row] * dinv[row]*dinv[col]
__global__ void k_scatter(const float* __restrict__ h, const float* __restrict__ dinv,
                          const int* __restrict__ row, const int* __restrict__ col,
                          float* __restrict__ out)
{
    const int CH = HH / 4;
    long gid = (long)blockIdx.x * blockDim.x + threadIdx.x;
    if (gid >= (long)EE * CH) return;
    long e  = gid / CH;
    int  c4 = (int)(gid - e * CH) * 4;
    int  r  = row[e];
    int  c  = col[e];
    float nrm = dinv[r] * dinv[c];
    const float4 hv = *(const float4*)&h[(long)r * HH + c4];
    float* dst = &out[(long)c * HH + c4];
    atomicAdd(dst + 0, hv.x * nrm);
    atomicAdd(dst + 1, hv.y * nrm);
    atomicAdd(dst + 2, hv.z * nrm);
    atomicAdd(dst + 3, hv.w * nrm);
}

// ---------------- BatchNorm (training stats) + LeakyReLU ----------------
__global__ void k_bn_stats(const float* __restrict__ x, float* __restrict__ stats)
{
    int  c  = blockIdx.x * blockDim.x + threadIdx.x;   // column, coalesced
    long r0 = (long)blockIdx.y * 256;
    long r1 = r0 + 256; if (r1 > NN) r1 = NN;
    float s = 0.0f, sq = 0.0f;
    for (long r = r0; r < r1; ++r) {
        float v = x[r * HH + c];
        s += v; sq += v * v;
    }
    atomicAdd(&stats[c], s);
    atomicAdd(&stats[HH + c], sq);
}

__global__ void k_bn_lrelu(float* __restrict__ x, const float* __restrict__ stats,
                           const float* __restrict__ g, const float* __restrict__ beta)
{
    long g4 = (long)blockIdx.x * blockDim.x + threadIdx.x;
    if (g4 >= (long)NN * HH / 4) return;
    long base = g4 * 4;
    int  c    = (int)(base % HH);
    const float inv_n = 1.0f / (float)NN;
    float4 sm = *(const float4*)&stats[c];          // column sums
    float4 sq = *(const float4*)&stats[HH + c];     // column sumsq
    float4 gg = *(const float4*)&g[c];
    float4 bb = *(const float4*)&beta[c];
    float4 xv = *(const float4*)&x[base];
    float4 o;
    {
        float mu = sm.x * inv_n, var = sq.x * inv_n - mu * mu;
        float v = gg.x * (xv.x - mu) * rsqrtf(var + EPS_) + bb.x;
        o.x = (v > 0.0f) ? v : v * SLOPE_;
    }
    {
        float mu = sm.y * inv_n, var = sq.y * inv_n - mu * mu;
        float v = gg.y * (xv.y - mu) * rsqrtf(var + EPS_) + bb.y;
        o.y = (v > 0.0f) ? v : v * SLOPE_;
    }
    {
        float mu = sm.z * inv_n, var = sq.z * inv_n - mu * mu;
        float v = gg.z * (xv.z - mu) * rsqrtf(var + EPS_) + bb.z;
        o.z = (v > 0.0f) ? v : v * SLOPE_;
    }
    {
        float mu = sm.w * inv_n, var = sq.w * inv_n - mu * mu;
        float v = gg.w * (xv.w - mu) * rsqrtf(var + EPS_) + bb.w;
        o.w = (v > 0.0f) ? v : v * SLOPE_;
    }
    *(float4*)&x[base] = o;
}

// ---------------- segment-max pooling ----------------
__device__ inline void atomicMaxFloat(float* addr, float val) {
    if (val >= 0.0f) atomicMax((int*)addr, __float_as_int(val));
    else             atomicMin((unsigned int*)addr, __float_as_uint(val));
}

__global__ void k_pool_max(const float* __restrict__ x, const int* __restrict__ batch,
                           float* __restrict__ pooled)
{
    long g4 = (long)blockIdx.x * blockDim.x + threadIdx.x;
    if (g4 >= (long)NN * HH / 4) return;
    long base = g4 * 4;
    long node = base / HH;
    int  c    = (int)(base - node * HH);
    float4 v  = *(const float4*)&x[base];
    float* dst = &pooled[(long)batch[node] * HH + c];
    atomicMaxFloat(dst + 0, v.x);
    atomicMaxFloat(dst + 1, v.y);
    atomicMaxFloat(dst + 2, v.z);
    atomicMaxFloat(dst + 3, v.w);
}

// ---------------- host launcher ----------------
static inline size_t alignup(size_t x) { return (x + 255) & ~(size_t)255; }

extern "C" void kernel_launch(void* const* d_in, const int* in_sizes, int n_in,
                              void* d_out, int out_size, void* d_ws, size_t ws_size,
                              hipStream_t stream) {
    const float* x     = (const float*)d_in[0];
    const int*   eidx  = (const int*)d_in[1];
    const int*   batch = (const int*)d_in[2];
    const float* W1    = (const float*)d_in[3];
    const float* b1    = (const float*)d_in[4];
    const float* g1    = (const float*)d_in[5];
    const float* bt1   = (const float*)d_in[6];
    const float* W2    = (const float*)d_in[7];
    const float* b2    = (const float*)d_in[8];
    const float* g2    = (const float*)d_in[9];
    const float* bt2   = (const float*)d_in[10];
    const float* Wf    = (const float*)d_in[11];
    const float* bf    = (const float*)d_in[12];
    float* out = (float*)d_out;

    const int* erow = eidx;
    const int* ecol = eidx + EE;

    // workspace carve-up
    char* ws = (char*)d_ws;
    size_t off = 0;
    float* dinv   = (float*)(ws + off); off += alignup((size_t)NN * 4);
    float* stats  = (float*)(ws + off); off += alignup((size_t)4 * HH * 4);   // [sum1,sq1,sum2,sq2]
    float* buf1   = (float*)(ws + off); off += alignup((size_t)NN * HH * 4);
    float* buf2   = (float*)(ws + off); off += alignup((size_t)NN * HH * 4);
    float* pooled = (float*)(ws + off); off += alignup((size_t)GG * HH * 4);

    const int T = 256;
    const long nh4 = (long)NN * HH / 4;   // float4 work-items for N x H passes

    // init
    k_init_deg   <<<(NN + T - 1) / T, T, 0, stream>>>(dinv);
    k_zero_stats <<<(4 * HH + T - 1) / T, T, 0, stream>>>(stats);
    k_init_pooled<<<((long)GG * HH + T - 1) / T, T, 0, stream>>>(pooled);

    // degrees -> dinv
    k_deg_edges<<<(EE + T - 1) / T, T, 0, stream>>>(ecol, dinv);
    k_dinv     <<<(NN + T - 1) / T, T, 0, stream>>>(dinv);

    // ---- layer 1 ----
    // buf1 = x @ W1   grid = (M/32) * (Nc/256) = 3125 * 2
    k_gemm<<<(NN / 32) * (HH / 256), T, 0, stream>>>(x, W1, nullptr, buf1, NN, DIN_, HH, 0);
    k_selfloop_bias<<<(nh4 + T - 1) / T, T, 0, stream>>>(buf1, dinv, b1, buf2);
    k_scatter<<<((long)EE * (HH / 4) + T - 1) / T, T, 0, stream>>>(buf1, dinv, erow, ecol, buf2);
    {
        dim3 g(HH / T, (NN + 255) / 256);
        k_bn_stats<<<g, T, 0, stream>>>(buf2, stats);
    }
    k_bn_lrelu<<<(nh4 + T - 1) / T, T, 0, stream>>>(buf2, stats, g1, bt1);

    // ---- layer 2 ----
    k_gemm<<<(NN / 32) * (HH / 256), T, 0, stream>>>(buf2, W2, nullptr, buf1, NN, HH, HH, 0);
    k_selfloop_bias<<<(nh4 + T - 1) / T, T, 0, stream>>>(buf1, dinv, b2, buf2);
    k_scatter<<<((long)EE * (HH / 4) + T - 1) / T, T, 0, stream>>>(buf1, dinv, erow, ecol, buf2);
    {
        dim3 g(HH / T, (NN + 255) / 256);
        k_bn_stats<<<g, T, 0, stream>>>(buf2, stats + 2 * HH);
    }
    k_bn_lrelu<<<(nh4 + T - 1) / T, T, 0, stream>>>(buf2, stats + 2 * HH, g2, bt2);

    // ---- pool + final linear ----
    k_pool_max<<<(nh4 + T - 1) / T, T, 0, stream>>>(buf2, batch, pooled);
    // out = sanitize(pooled) @ Wf + bf   grid = (4096/32) * (256/256) = 128
    k_gemm<<<(GG / 32) * (OUTC / 256), T, 0, stream>>>(pooled, Wf, bf, out, GG, HH, OUTC, 1);
}
